// BasicUpdateBlockDepth_7576322310903
// MI455X (gfx1250) — matrix operand order repaired and checked by reading, not compiled, not run on hardware
//
#include <hip/hip_runtime.h>
#include <hip/hip_bf16.h>
#include <math.h>

// ---------------------------------------------------------------------------
// Problem constants (from reference): compile-time since device scalars are
// unreadable on host during graph capture.
// ---------------------------------------------------------------------------
#define BATCH 2
#define IMG_H 120
#define IMG_W 160
#define HID   128
#define CTX   192
#define BINS  16
#define SEQL  3
#define MIN_D 0.001f
#define MAX_D 10.0f

typedef __attribute__((ext_vector_type(16))) _Float16 v16h;
typedef __attribute__((ext_vector_type(8)))  _Float16 v8h;
typedef __attribute__((ext_vector_type(8)))  float    v8f;

// Async global->LDS copies (CDNA5, ASYNCcnt-tracked). Guarded so the file
// still compiles on toolchains without the builtin. The builtin's pointer
// params are typed as int4* in AS(1)/AS(3) (per hipcc diagnostic).
#if defined(__has_builtin)
#if __has_builtin(__builtin_amdgcn_global_load_async_to_lds_b128) && \
    __has_builtin(__builtin_amdgcn_s_wait_asynccnt)
#define USE_ASYNC_LDS 1
#endif
#endif
#ifndef USE_ASYNC_LDS
#define USE_ASYNC_LDS 0
#endif

#if USE_ASYNC_LDS
typedef int async_b128_t __attribute__((vector_size(4 * sizeof(int))));
#define ASYNC_COPY_B128(gsrc, ldst)                                          \
    __builtin_amdgcn_global_load_async_to_lds_b128(                          \
        (__attribute__((address_space(1))) async_b128_t*)(gsrc),             \
        (__attribute__((address_space(3))) async_b128_t*)(ldst), 0, 0)
#endif

// ---------------------------------------------------------------------------
// Implicit-GEMM convolution, WMMA f16 (f32 accumulate), register-tiled.
//   out[b, co, p] = act( bias[co] + sum_k wgt16[co, k] * im2col(in)[k, p] )
// M = Cout, N = H*W pixels, K = Cin*KH*KW. Weights pre-converted to f16.
// Block tile: 128(M) x 64(N); 8 wave32 in a 4(M) x 2(N) grid; each wave owns
// a 2x2 grid of 16x16 accumulators -> 4 v_wmma_f32_16x16x32_f16 per K-step
// from 2 A-fragments x 2 B-fragments. K stepped by 32.
// A tile: contiguous f16 rows -> async b128 DMA to LDS (OOB -> zero guard).
// B tile: im2col gather, stored transposed [n][k] so fragments are two
// aligned b128 DS loads per lane.
// ACT: 0=none 1=relu 2=sigmoid 3=tanh
// ---------------------------------------------------------------------------
#define MT 128
#define NT 64
#define KT 32
#define A_STR 40   // halves per LDS row (80B: 16B aligned, bank-skewed)
#define B_STR 40

template <int ACT>
__global__ void __launch_bounds__(256)
conv_wmma(const float* __restrict__ in, const _Float16* __restrict__ wgt16,
          const _Float16* __restrict__ guard64, const float* __restrict__ bias,
          float* __restrict__ out,
          int Cin, int Cout, int KH, int KW, int padH, int padW)
{
    __shared__ __align__(16) _Float16 lA[MT * A_STR]; // weights  [m][k]
    __shared__ __align__(16) _Float16 lB[NT * B_STR]; // im2col^T [n][k]

    const int HWp  = IMG_H * IMG_W;
    const int tid  = threadIdx.x;
    const int lane = tid & 31;
    const int wave = tid >> 5;
    const int waveM = wave >> 1;  // 0..3 -> 32 output channels each
    const int waveN = wave & 1;   // 0..1 -> 32 pixels each

    const int nTile  = blockIdx.x * NT;
    const int coTile = blockIdx.y * MT;
    const int b      = blockIdx.z;

    const int KHW  = KH * KW;
    const int Ktot = Cin * KHW;

    const float* inB = in + (size_t)b * Cin * HWp;

    v8f acc[2][2] = {};

    // A staging: 128 rows x 32 halves = 512 b128 chunks (8 halves each),
    // 2 chunks per thread. chunk c: row m = c>>2, kOff = (c&3)*8.
    const int ac0 = tid * 2;
    // B staging: one k per thread (k = tid&31), 8 pixels each.
    const int bk  = tid & 31;
    const int bn0 = (tid >> 5) * 8;

    for (int k0 = 0; k0 < Ktot; k0 += KT) {
        // ---- stage A: f16 weight tile [MT x KT] ----
        #pragma unroll
        for (int i = 0; i < 2; ++i) {
            const int c    = ac0 + i;
            const int m    = c >> 2;
            const int kOff = (c & 3) * 8;
            const int co   = coTile + m;
            const bool ok  = (co < Cout) && (k0 + kOff + 8 <= Ktot);
            const _Float16* gsrc =
                ok ? (wgt16 + (size_t)co * Ktot + k0 + kOff) : guard64;
            _Float16* ldst = &lA[m * A_STR + kOff];
#if USE_ASYNC_LDS
            ASYNC_COPY_B128(gsrc, ldst);
#else
            *(v8h*)ldst = *(const v8h*)gsrc;
#endif
        }
        // ---- stage B: im2col tile, transposed lB[n][k], f32 -> f16 ----
        {
            const int kk = k0 + bk;
            int ci = 0, kh = 0, kw = 0;
            const bool kOK = kk < Ktot;
            if (kOK) {
                ci = kk / KHW;
                const int r2 = kk - ci * KHW;
                kh = r2 / KW;
                kw = r2 - kh * KW;
            }
            const float* src = inB + (size_t)ci * HWp;
            if (kOK) __builtin_prefetch(src + nTile, 0, 1); // global_prefetch_b8
            #pragma unroll
            for (int j = 0; j < 8; ++j) {
                const int nSub = bn0 + j;
                const int pix  = nTile + nSub;
                float v = 0.0f;
                if (kOK && pix < HWp) {
                    const int y  = pix / IMG_W;
                    const int x  = pix - y * IMG_W;
                    const int iy = y + kh - padH;
                    const int ix = x + kw - padW;
                    if (iy >= 0 && iy < IMG_H && ix >= 0 && ix < IMG_W)
                        v = src[iy * IMG_W + ix];
                }
                lB[nSub * B_STR + bk] = (_Float16)v;
            }
        }
#if USE_ASYNC_LDS
        __builtin_amdgcn_s_wait_asynccnt(0);
#endif
        __syncthreads();

        // ---- fragments per documented 16-bit WMMA VGPR layouts ----
        // A 16x32: lanes 0-15 -> rows; lane>>4 picks K half kb in {0,8};
        //          VGPR0-3 = K[kb..kb+7], VGPR4-7 = K[16+kb..16+kb+7].
        // B 32x16: lane&15 -> column; same K split as A (kb..kb+7, 16+kb..16+kb+7).
        v16h afrag[2], bfrag[2];
        {
            const int akb = (lane >> 4) * 8;
            const int bkb = (lane >> 4) * 8;
            #pragma unroll
            for (int mi = 0; mi < 2; ++mi) {
                const int arow = waveM * 32 + mi * 16 + (lane & 15);
                const v8h a_lo = *(const v8h*)&lA[arow * A_STR + akb];
                const v8h a_hi = *(const v8h*)&lA[arow * A_STR + 16 + akb];
                #pragma unroll
                for (int j = 0; j < 8; ++j) {
                    afrag[mi][j] = a_lo[j]; afrag[mi][8 + j] = a_hi[j];
                }
            }
            #pragma unroll
            for (int ni = 0; ni < 2; ++ni) {
                const int bcol = waveN * 32 + ni * 16 + (lane & 15);
                const v8h b_lo = *(const v8h*)&lB[bcol * B_STR + bkb];
                const v8h b_hi = *(const v8h*)&lB[bcol * B_STR + 16 + bkb];
                #pragma unroll
                for (int j = 0; j < 8; ++j) {
                    bfrag[ni][j] = b_lo[j]; bfrag[ni][8 + j] = b_hi[j];
                }
            }
        }
        #pragma unroll
        for (int mi = 0; mi < 2; ++mi)
            #pragma unroll
            for (int ni = 0; ni < 2; ++ni)
                acc[mi][ni] = __builtin_amdgcn_wmma_f32_16x16x32_f16(
                    false, afrag[mi], false, bfrag[ni], (short)0,
                    acc[mi][ni], false, false);
        __syncthreads();
    }

    // ---- epilogue: bias + activation + store
    //      (C/D layout: VGPR v -> M = 8*(lane>>4)+v, N = lane&15) ----
    #pragma unroll
    for (int mi = 0; mi < 2; ++mi) {
        const int co0 = coTile + waveM * 32 + mi * 16 + (lane >> 4) * 8;
        #pragma unroll
        for (int ni = 0; ni < 2; ++ni) {
            const int pix = nTile + waveN * 32 + ni * 16 + (lane & 15);
            if (pix < HWp) {
                #pragma unroll
                for (int v = 0; v < 8; ++v) {
                    const int co = co0 + v;
                    if (co < Cout) {
                        float r = acc[mi][ni][v] + bias[co];
                        if (ACT == 1)      r = fmaxf(r, 0.0f);
                        else if (ACT == 2) r = 1.0f / (1.0f + __expf(-r));
                        else if (ACT == 3) r = tanhf(r);
                        out[((size_t)b * Cout + co) * HWp + pix] = r;
                    }
                }
            }
        }
    }
}

// ---------------------------------------------------------------------------
// One-time weight conversion f32 -> f16 (weights reused by ~1800 tiles).
// ---------------------------------------------------------------------------
__global__ void cvt_w_f16(_Float16* __restrict__ dst, const float* __restrict__ src, int n)
{
    const int i = blockIdx.x * blockDim.x + threadIdx.x;
    if (i < n) dst[i] = (_Float16)src[i];
}

__global__ void zero_guard(_Float16* __restrict__ g)
{
    if (threadIdx.x < 32) g[threadIdx.x] = (_Float16)0.0f;
}

// ---------------------------------------------------------------------------
// Elementwise helpers
// ---------------------------------------------------------------------------
__global__ void init_bins(float* __restrict__ edges, float* __restrict__ curr)
{
    const int HWp = IMG_H * IMG_W;
    const int idx = blockIdx.x * blockDim.x + threadIdx.x;
    if (idx >= BATCH * HWp) return;
    const int b = idx / HWp, p = idx - b * HWp;
    const float step = (MAX_D - MIN_D) / (float)BINS;
    #pragma unroll
    for (int j = 0; j <= BINS; ++j)
        edges[((size_t)b * (BINS + 1) + j) * HWp + p] = MIN_D + j * step;
    #pragma unroll
    for (int j = 0; j < BINS; ++j)
        curr[((size_t)b * BINS + j) * HWp + p] = MIN_D + (j + 0.5f) * step;
}

// dst[b, chanOff + c, p] = src[b, c, p]; src is [B, nCh, HW]
__global__ void copy_channels(float* __restrict__ dst, const float* __restrict__ src,
                              int chanOff, int nCh, int CHdst, int n)
{
    const int HWp = IMG_H * IMG_W;
    const int i = blockIdx.x * blockDim.x + threadIdx.x;
    if (i >= n) return;
    const int per = nCh * HWp;
    const int b = i / per, r = i - b * per;
    dst[((size_t)b * CHdst + chanOff) * HWp + r] = src[i];
}

// hx[b, 0..HID, p] = r[b,c,p] * h[b,c,p]
__global__ void rh_into_hx(float* __restrict__ hx, const float* __restrict__ r,
                           const float* __restrict__ h, int CHdst, int n)
{
    const int HWp = IMG_H * IMG_W;
    const int i = blockIdx.x * blockDim.x + threadIdx.x;
    if (i >= n) return;
    const int per = HID * HWp;
    const int b = i / per, rr = i - b * per;
    hx[(size_t)b * CHdst * HWp + rr] = r[i] * h[i];
}

// h = (1-z)*h + z*q
__global__ void h_update(float* __restrict__ h, const float* __restrict__ z,
                         const float* __restrict__ q, int n)
{
    const int i = blockIdx.x * blockDim.x + threadIdx.x;
    if (i >= n) return;
    const float zi = z[i];
    h[i] = (1.0f - zi) * h[i] + zi * q[i];
}

// ---------------------------------------------------------------------------
// Per-pixel: softmax over 16 logits, depth_r, unc, label/depth_c, write the
// three stacked outputs for this iteration, and regenerate edges/curr.
// ---------------------------------------------------------------------------
__global__ void finalize_iter(const float* __restrict__ logits,
                              float* __restrict__ curr,
                              float* __restrict__ edges,
                              float* __restrict__ outBase, int it)
{
    const int HWp = IMG_H * IMG_W;
    const int idx = blockIdx.x * blockDim.x + threadIdx.x;
    if (idx >= BATCH * HWp) return;
    const int b = idx / HWp, p = idx - b * HWp;

    const float* lg = logits + (size_t)b * BINS * HWp + p;
    float*       cu = curr   + (size_t)b * BINS * HWp + p;
    float*       ed = edges  + (size_t)b * (BINS + 1) * HWp + p;

    float l[BINS], c[BINS];
    float m = -3.402823e38f;
    #pragma unroll
    for (int j = 0; j < BINS; ++j) { l[j] = lg[j * HWp]; m = fmaxf(m, l[j]); }
    float s = 0.0f;
    #pragma unroll
    for (int j = 0; j < BINS; ++j) { l[j] = __expf(l[j] - m); s += l[j]; }
    const float inv = 1.0f / s;

    float dr = 0.0f;
    #pragma unroll
    for (int j = 0; j < BINS; ++j) {
        c[j] = cu[j * HWp];
        l[j] *= inv;                 // prob
        dr += l[j] * c[j];
    }
    float var = 0.0f;
    #pragma unroll
    for (int j = 0; j < BINS; ++j) {
        const float d = c[j] - dr;
        var += l[j] * d * d;
    }
    const float un = sqrtf(fmaxf(var, 0.0f));

    // half-open bin membership against current edges
    int label = 0;
    float ePrev = ed[0];
    #pragma unroll
    for (int j = 0; j < BINS; ++j) {
        const float eNext = ed[(j + 1) * HWp];
        if (dr >= ePrev && dr < eNext) label = j;
        ePrev = eNext;
    }
    const float dc = c[label];

    // outputs: (3,B,1,H,W) for r, then c, then u, concatenated flat.
    const size_t ob  = (size_t)it * BATCH * HWp + (size_t)b * HWp + p;
    const size_t st3 = (size_t)SEQL * BATCH * HWp;
    outBase[ob]           = dr;
    outBase[st3 + ob]     = dc;
    outBase[2 * st3 + ob] = un;

    // update_sample ('direct'): edges = clip(start + j*step), curr = midpoints
    const float start = fmaxf(dr - 0.5f * un, MIN_D);
    const float step  = un * (1.0f / (float)BINS);
    float prev = 0.0f;
    #pragma unroll
    for (int j = 0; j <= BINS; ++j) {
        const float ej = fminf(fmaxf(start + j * step, MIN_D), MAX_D);
        ed[j * HWp] = ej;
        if (j > 0) cu[(j - 1) * HWp] = 0.5f * (prev + ej);
        prev = ej;
    }
}

// ---------------------------------------------------------------------------
// Host-side orchestration
// ---------------------------------------------------------------------------
static inline void launch_conv(int act, const float* in, const _Float16* w16,
                               const _Float16* guard, const float* bias, float* out,
                               int Cin, int Cout, int KH, int KW,
                               int padH, int padW, hipStream_t s)
{
    const int HWp = IMG_H * IMG_W;
    dim3 grid((HWp + NT - 1) / NT, (Cout + MT - 1) / MT, BATCH);
    dim3 block(256);
    switch (act) {
    case 0: conv_wmma<0><<<grid, block, 0, s>>>(in, w16, guard, bias, out, Cin, Cout, KH, KW, padH, padW); break;
    case 1: conv_wmma<1><<<grid, block, 0, s>>>(in, w16, guard, bias, out, Cin, Cout, KH, KW, padH, padW); break;
    case 2: conv_wmma<2><<<grid, block, 0, s>>>(in, w16, guard, bias, out, Cin, Cout, KH, KW, padH, padW); break;
    default: conv_wmma<3><<<grid, block, 0, s>>>(in, w16, guard, bias, out, Cin, Cout, KH, KW, padH, padW); break;
    }
}

extern "C" void kernel_launch(void* const* d_in, const int* in_sizes, int n_in,
                              void* d_out, int out_size, void* d_ws, size_t ws_size,
                              hipStream_t stream)
{
    (void)in_sizes; (void)n_in; (void)out_size; (void)ws_size;

    const int HWp = IMG_H * IMG_W;
    const int BHW = BATCH * HWp;
    const int GIN = HID + 2 * HID + CTX;   // 576

    // Inputs: depth(shape-only), context, gru_hidden, then params in insertion
    // order (sorted keys, _w before _b), then scalars (seq_len etc, unused).
    const float* context = (const float*)d_in[1];
    const float* gruh    = (const float*)d_in[2];
    const float *ed1_w = (const float*)d_in[3],  *ed1_b = (const float*)d_in[4];
    const float *ed2_w = (const float*)d_in[5],  *ed2_b = (const float*)d_in[6];
    const float *ed3_w = (const float*)d_in[7],  *ed3_b = (const float*)d_in[8];
    const float *ed4_w = (const float*)d_in[9],  *ed4_b = (const float*)d_in[10];
    const float *gq1_w = (const float*)d_in[11], *gq1_b = (const float*)d_in[12];
    const float *gq2_w = (const float*)d_in[13], *gq2_b = (const float*)d_in[14];
    const float *gr1_w = (const float*)d_in[15], *gr1_b = (const float*)d_in[16];
    const float *gr2_w = (const float*)d_in[17], *gr2_b = (const float*)d_in[18];
    const float *gz1_w = (const float*)d_in[19], *gz1_b = (const float*)d_in[20];
    const float *gz2_w = (const float*)d_in[21], *gz2_b = (const float*)d_in[22];
    const float *p1_w  = (const float*)d_in[23], *p1_b  = (const float*)d_in[24];
    const float *p2_w  = (const float*)d_in[25], *p2_b  = (const float*)d_in[26];

    float* outp = (float*)d_out;

    // Workspace carve (f32 region)
    float* ws = (float*)d_ws;
    float* curr   = ws; ws += (size_t)BINS * BHW;
    float* edges  = ws; ws += (size_t)(BINS + 1) * BHW;
    float* t1     = ws; ws += (size_t)HID * BHW;       // encoder stage 1 / p1 out
    float* t2     = ws; ws += (size_t)HID * BHW;       // encoder stage 2
    float* t3     = ws; ws += (size_t)(2 * HID) * BHW; // encoder stage 3
    float* feat   = ws; ws += (size_t)(2 * HID) * BHW; // encoder out
    float* hx     = ws; ws += (size_t)GIN * BHW;       // [h | feat | context]
    float* zb     = ws; ws += (size_t)HID * BHW;
    float* rb     = ws; ws += (size_t)HID * BHW;
    float* qb     = ws; ws += (size_t)HID * BHW;
    float* hb     = ws; ws += (size_t)HID * BHW;
    float* logits = ws; ws += (size_t)BINS * BHW;

    // f16 weight arena + zero guard (aligned: everything above is 4B units)
    _Float16* wh = (_Float16*)ws;
    const int KED1 = BINS * 49, K33H = HID * 9, K33F = 2 * HID * 9, KG = GIN * 5;
    struct { const float* src; int n; } wlist[12] = {
        { ed1_w, HID * KED1 },     { ed2_w, HID * K33H },
        { ed3_w, 2 * HID * K33H }, { ed4_w, 2 * HID * K33F },
        { gz1_w, HID * KG }, { gr1_w, HID * KG }, { gq1_w, HID * KG },
        { gz2_w, HID * KG }, { gr2_w, HID * KG }, { gq2_w, HID * KG },
        { p1_w,  HID * K33H },     { p2_w,  BINS * K33H } };
    _Float16* w16[12];
    {
        _Float16* cur = wh;
        for (int i = 0; i < 12; ++i) { w16[i] = cur; cur += wlist[i].n; }
        // 16B-align the guard block
        cur = (_Float16*)(((size_t)(cur) + 15) & ~(size_t)15);
        wh = cur;
    }
    _Float16* guard = wh;  // 64B zero block for OOB async redirects

    const int T = 256;
    const int nPix  = BHW;
    const int nHid  = HID * BHW;
    const int nFeat = 2 * HID * BHW;
    const int nCtx  = CTX * BHW;

    // ---- one-time prep: f16 weights, guard, bins, h copy, context in hx ----
    for (int i = 0; i < 12; ++i)
        cvt_w_f16<<<(wlist[i].n + T - 1) / T, T, 0, stream>>>(w16[i], wlist[i].src, wlist[i].n);
    zero_guard<<<1, 32, 0, stream>>>(guard);
    init_bins<<<(nPix + T - 1) / T, T, 0, stream>>>(edges, curr);
    copy_channels<<<(nHid + T - 1) / T, T, 0, stream>>>(hb, gruh, 0, HID, HID, nHid);
    copy_channels<<<(nCtx + T - 1) / T, T, 0, stream>>>(hx, context, HID + 2 * HID, CTX, GIN, nCtx);

    for (int it = 0; it < SEQL; ++it) {
        // ---- encoder on sg(curr): 4 relu convs ----
        launch_conv(1, curr, w16[0], guard, ed1_b, t1,   BINS,    HID,     7, 7, 3, 3, stream);
        launch_conv(1, t1,   w16[1], guard, ed2_b, t2,   HID,     HID,     3, 3, 1, 1, stream);
        launch_conv(1, t2,   w16[2], guard, ed3_b, t3,   HID,     2 * HID, 3, 3, 1, 1, stream);
        launch_conv(1, t3,   w16[3], guard, ed4_b, feat, 2 * HID, 2 * HID, 3, 3, 1, 1, stream);

        // hx = [h | feat | context]
        copy_channels<<<(nHid  + T - 1) / T, T, 0, stream>>>(hx, hb,   0,   HID,     GIN, nHid);
        copy_channels<<<(nFeat + T - 1) / T, T, 0, stream>>>(hx, feat, HID, 2 * HID, GIN, nFeat);

        // ---- GRU pass 1: (1,5) kernels, pad (0,2) ----
        launch_conv(2, hx, w16[4], guard, gz1_b, zb, GIN, HID, 1, 5, 0, 2, stream);
        launch_conv(2, hx, w16[5], guard, gr1_b, rb, GIN, HID, 1, 5, 0, 2, stream);
        rh_into_hx<<<(nHid + T - 1) / T, T, 0, stream>>>(hx, rb, hb, GIN, nHid);
        launch_conv(3, hx, w16[6], guard, gq1_b, qb, GIN, HID, 1, 5, 0, 2, stream);
        h_update<<<(nHid + T - 1) / T, T, 0, stream>>>(hb, zb, qb, nHid);
        copy_channels<<<(nHid + T - 1) / T, T, 0, stream>>>(hx, hb, 0, HID, GIN, nHid);

        // ---- GRU pass 2: (5,1) kernels, pad (2,0) ----
        launch_conv(2, hx, w16[7], guard, gz2_b, zb, GIN, HID, 5, 1, 2, 0, stream);
        launch_conv(2, hx, w16[8], guard, gr2_b, rb, GIN, HID, 5, 1, 2, 0, stream);
        rh_into_hx<<<(nHid + T - 1) / T, T, 0, stream>>>(hx, rb, hb, GIN, nHid);
        launch_conv(3, hx, w16[9], guard, gq2_b, qb, GIN, HID, 5, 1, 2, 0, stream);
        h_update<<<(nHid + T - 1) / T, T, 0, stream>>>(hb, zb, qb, nHid);

        // ---- probability head ----
        launch_conv(1, hb, w16[10], guard, p1_b, t1,     HID, HID,  3, 3, 1, 1, stream);
        launch_conv(0, t1, w16[11], guard, p2_b, logits, HID, BINS, 3, 3, 1, 1, stream);

        // ---- softmax + stats + outputs + re-binning ----
        finalize_iter<<<(nPix + T - 1) / T, T, 0, stream>>>(logits, curr, edges, outp, it);
    }
}
